// RelativeMultiHeadAttention_56745107915289
// MI455X (gfx1250) — matrix operand order repaired and checked by reading, not compiled, not run on hardware
//
#include <hip/hip_runtime.h>

// ---------------------------------------------------------------------------
// Relative multi-head attention forward for MI455X (gfx1250, wave32, WMMA).
// T=1024, B=4, E=1024, H=16, hd=64.
// ---------------------------------------------------------------------------

typedef __bf16 bf16_t;
typedef __attribute__((ext_vector_type(16))) __bf16 v16bf;
typedef __attribute__((ext_vector_type(2)))  __bf16 v2bf;
typedef __attribute__((ext_vector_type(8)))  float  v8f;
typedef __attribute__((__vector_size__(16))) int    v4i_t;   // b128 payload

#define T_SEQ 1024
#define BATCH 4
#define EMB   1024
#define HEADS 16
#define HD    64
#define NHEAD (BATCH * HEADS)   // 64

#if __has_builtin(__builtin_amdgcn_global_load_async_to_lds_b128) && \
    __has_builtin(__builtin_amdgcn_s_wait_asynccnt)
#define HAVE_ASYNC_LDS 1
#else
#define HAVE_ASYNC_LDS 0
#endif

union Frag16 {
    v16bf v;
    unsigned u[8];
    uint4 q[2];
};

// two f32 -> packed bf16 dword; prefer the native v_cvt_pk_bf16_f32 path
__device__ __forceinline__ unsigned pack_bf16(float lo, float hi) {
#if __has_builtin(__builtin_amdgcn_cvt_pk_bf16_f32)
    v2bf r = __builtin_amdgcn_cvt_pk_bf16_f32(lo, hi);
    return __builtin_bit_cast(unsigned, r);
#else
    union { __bf16 h[2]; unsigned u; } x;
    x.h[0] = (__bf16)lo; x.h[1] = (__bf16)hi;
    return x.u;
#endif
}

__device__ __forceinline__ v8f wmma_bf16(v16bf a, v16bf b, v8f c) {
    // D = A(16x32 bf16) * B(32x16 bf16) + C(16x16 f32)
    return __builtin_amdgcn_wmma_f32_16x16x32_bf16(
        false, a, false, b, (short)0, c, false, false);
}

// ---- fragment loaders (all-bf16 sources, pure b128 loads) ------------------
// A 16x32 bf16, this code's contiguous k order: lanes 0-15: M=lane, K=k0..k0+15
//                             lanes16-31: M=lane-16, K=k0+16..k0+31
__device__ __forceinline__ v16bf load_a_bf16(const bf16_t* base, int ld,
                                             int row0, int k0, int lane) {
    const int half = lane >> 4, lr = lane & 15;
    const bf16_t* p = base + (size_t)(row0 + lr) * ld + k0 + half * 16;
    Frag16 f;
    f.q[0] = *(const uint4*)(p);
    f.q[1] = *(const uint4*)(p + 8);
    return f.v;
}
// B 32x16 bf16, this code's contiguous k order: lanes 0-15: N=lane, K=k0..k0+15 (pairs/dword),
//                             lanes16-31: N=lane-16, K=k0+16..k0+31.
// Source is row-major [N][K] (i.e. B[k][n] = src[n][k]).
__device__ __forceinline__ v16bf load_b_bf16(const bf16_t* base, int ld,
                                             int n0, int k0, int lane) {
    const int half = lane >> 4, lr = lane & 15;
    const bf16_t* p = base + (size_t)(n0 + lr) * ld + k0 + half * 16;
    Frag16 f;
    f.q[0] = *(const uint4*)(p);
    f.q[1] = *(const uint4*)(p + 8);
    return f.v;
}

// ---------------------------------------------------------------------------
// Kernel 0: elementwise f32 -> bf16 (8 elements / thread, vector ld/st)
// ---------------------------------------------------------------------------
__global__ void __launch_bounds__(256)
cvt_bf16_kernel(const float* __restrict__ src, bf16_t* __restrict__ dst, int n8) {
    const int i = blockIdx.x * blockDim.x + threadIdx.x;
    if (i >= n8) return;
    const float4* s = (const float4*)src + (size_t)i * 2;
    float4 a = s[0], b = s[1];
    uint4 w;
    w.x = pack_bf16(a.x, a.y);
    w.y = pack_bf16(a.z, a.w);
    w.z = pack_bf16(b.x, b.y);
    w.w = pack_bf16(b.z, b.w);
    ((uint4*)dst)[i] = w;
}

// ---------------------------------------------------------------------------
// Kernel 1: fused QKV projection. C[m,f] = X[m,:] . W[f,:] + bias[f]
// M=4096 (m = t*B+b), N=3072, K=1024.  One wave owns 16 rows x 64 cols.
// Scatter bf16 results head-major: Q[n,t,d]*0.125, K[n,s,d], Vt[n,d,t].
// ---------------------------------------------------------------------------
__global__ void __launch_bounds__(256)
qkv_proj_kernel(const bf16_t* __restrict__ X, const bf16_t* __restrict__ W,
                const float* __restrict__ bias,
                bf16_t* __restrict__ Qb, bf16_t* __restrict__ Kb,
                bf16_t* __restrict__ Vt) {
    const int lane = threadIdx.x & 31;
    const int wid  = blockIdx.x * (blockDim.x >> 5) + (threadIdx.x >> 5);
    const int mtile = wid / 48;       // 0..255
    const int ngrp  = wid % 48;       // 0..47
    const int m0 = mtile * 16;
    const int f0 = ngrp * 64;

    v8f acc[4] = {};
    for (int k0 = 0; k0 < EMB; k0 += 32) {
        v16bf a = load_a_bf16(X, EMB, m0, k0, lane);
#pragma unroll
        for (int j = 0; j < 4; ++j) {
            v16bf b = load_b_bf16(W, EMB, f0 + j * 16, k0, lane);
            acc[j] = wmma_bf16(a, b, acc[j]);
        }
    }

    const int half = lane >> 4, lr = lane & 15;
#pragma unroll
    for (int j = 0; j < 4; ++j) {
        const int f  = f0 + j * 16 + lr;
        const float bv = bias[f];
        const int sec = f >> 10;          // 0=Q 1=K 2=V
        const int fi  = f & 1023;
        const int h   = fi >> 6;
        const int d   = fi & 63;
#pragma unroll
        for (int r = 0; r < 8; ++r) {
            const int m = m0 + half * 8 + r;
            const int t = m >> 2, bb = m & 3;
            const int n = bb * HEADS + h;
            const float val = acc[j][r] + bv;
            if (sec == 0)
                Qb[((size_t)n * T_SEQ + t) * HD + d] = (bf16_t)(val * 0.125f);
            else if (sec == 1)
                Kb[((size_t)n * T_SEQ + t) * HD + d] = (bf16_t)val;
            else
                Vt[((size_t)n * HD + d) * T_SEQ + t] = (bf16_t)val;
        }
    }
}

// ---------------------------------------------------------------------------
// Kernel 2: attention with online softmax. One wave per (head n, 16 query rows).
// Computes S^T = K . Q^T (M=s, N=t) so softmax rows live in the lane dim,
// then O^T = V^T . P^T with a lane-local (+one half-wave shuffle) C->B repack.
// The positions_bias HBM stream (256 MB) is double-buffered into LDS with
// GLOBAL_LOAD_ASYNC_TO_LDS_B128 + s_wait_asynccnt when the toolchain has it.
// ---------------------------------------------------------------------------

#if HAVE_ASYNC_LDS
// issue one 16x32 f32 bias tile (2 KB) into LDS: 4 x b128 per lane
__device__ __forceinline__ void bias_async_issue(const float* g, float* l,
                                                 int lane) {
#pragma unroll
    for (int i = 0; i < 4; ++i) {
        const int c   = lane + 32 * i;      // 0..127 chunk id (16B chunks)
        const int row = c >> 3;             // 0..15
        const int col = (c & 7) * 4;        // 0,4,..,28
        __builtin_amdgcn_global_load_async_to_lds_b128(
            (v4i_t*)(g + row * T_SEQ + col),
            (v4i_t*)(l + row * 32 + col),
            0, 0);
    }
}
#endif

__global__ void __launch_bounds__(32)
attention_kernel(const bf16_t* __restrict__ Qb, const bf16_t* __restrict__ Kb,
                 const bf16_t* __restrict__ Vt, const float* __restrict__ pbias,
                 const unsigned char* __restrict__ mask,
                 bf16_t* __restrict__ Obuf) {
    const int lane = threadIdx.x & 31;
    const int n  = blockIdx.x;            // 0..63  (b*H + h)
    const int t0 = blockIdx.y * 16;
    const int bb = n / HEADS;
    const int h  = n % HEADS;
    const int half = lane >> 4, lr = lane & 15;
    const int t = t0 + lr;

    const bf16_t* Qn = Qb + (size_t)n * T_SEQ * HD;
    const bf16_t* Kn = Kb + (size_t)n * T_SEQ * HD;
    const bf16_t* Vn = Vt + (size_t)n * HD * T_SEQ;
    const float*  Pn = pbias + (size_t)n * T_SEQ * T_SEQ;   // head's bias block
    const float*  Bn = Pn + (size_t)t * T_SEQ;              // per-lane row
    const unsigned char* Mb = mask + (size_t)bb * T_SEQ;

#if HAVE_ASYNC_LDS
    __shared__ float bsh[2][16 * 32];     // double-buffered 16x32 bias tiles
    bias_async_issue(Pn + (size_t)t0 * T_SEQ, bsh[0], lane);
#endif

    // B fragments of Q^T (K-dim = d), fixed across the whole s loop
    const v16bf bq0 = load_b_bf16(Qn, HD, t0, 0, lane);
    const v16bf bq1 = load_b_bf16(Qn, HD, t0, 32, lane);

    float m_run = -1e30f, l_run = 0.f;
    v8f o[4] = {};

    for (int s0 = 0; s0 < T_SEQ; s0 += 32) {
#if HAVE_ASYNC_LDS
        const int bufi = (s0 >> 5) & 1;
        if (s0 + 32 < T_SEQ) {
            // prefetch next tile into the other buffer, then wait for current:
            // async loads complete in order, so <=4 outstanding => current done
            bias_async_issue(Pn + (size_t)t0 * T_SEQ + s0 + 32, bsh[bufi ^ 1],
                             lane);
            __builtin_amdgcn_s_wait_asynccnt(4);
        } else {
            __builtin_amdgcn_s_wait_asynccnt(0);
        }
#else
        if (s0 + 128 < T_SEQ)
            __builtin_prefetch(Bn + s0 + 128, 0, 3);
#endif

        v8f st[2];
#pragma unroll
        for (int stile = 0; stile < 2; ++stile) {
            v16bf a0 = load_a_bf16(Kn, HD, s0 + stile * 16, 0, lane);
            v16bf a1 = load_a_bf16(Kn, HD, s0 + stile * 16, 32, lane);
            v8f z = {};
            z = wmma_bf16(a0, bq0, z);
            z = wmma_bf16(a1, bq1, z);
            // relative position bias: reg r <-> s = s0 + stile*16 + half*8 + r
#if HAVE_ASYNC_LDS
            const float* bp = &bsh[bufi][lr * 32 + stile * 16 + half * 8];
#else
            const float* bp = Bn + s0 + stile * 16 + half * 8;
#endif
            float4 b0 = *(const float4*)(bp);
            float4 b1 = *(const float4*)(bp + 4);
            z[0] += b0.x; z[1] += b0.y; z[2] += b0.z; z[3] += b0.w;
            z[4] += b1.x; z[5] += b1.y; z[6] += b1.z; z[7] += b1.w;
            // key padding mask
            unsigned long long mm =
                *(const unsigned long long*)(Mb + s0 + stile * 16 + half * 8);
#pragma unroll
            for (int r = 0; r < 8; ++r)
                if ((mm >> (8 * r)) & 0xffull) z[r] = -1e30f;
            st[stile] = z;
        }
        // online softmax over s (row = t = lane%16; both half-waves share t)
        float cmax = -1e30f;
#pragma unroll
        for (int r = 0; r < 8; ++r) cmax = fmaxf(cmax, fmaxf(st[0][r], st[1][r]));
        cmax = fmaxf(cmax, __shfl_xor(cmax, 16, 32));
        const float m_new = fmaxf(m_run, cmax);
        const float alpha = __expf(m_run - m_new);
        v8f p0, p1;
        float csum = 0.f;
#pragma unroll
        for (int r = 0; r < 8; ++r) {
            p0[r] = __expf(st[0][r] - m_new);
            p1[r] = __expf(st[1][r] - m_new);
            csum += p0[r] + p1[r];
        }
        csum += __shfl_xor(csum, 16, 32);
        l_run = l_run * alpha + csum;
        m_run = m_new;
#pragma unroll
        for (int i = 0; i < 4; ++i)
#pragma unroll
            for (int r = 0; r < 8; ++r) o[i][r] *= alpha;

        // C-layout (S^T) -> B-layout (P^T): half-wave swap + bf16 pack
        float p0x[8], p1x[8];
#pragma unroll
        for (int r = 0; r < 8; ++r) {
            p0x[r] = __shfl_xor(p0[r], 16, 32);
            p1x[r] = __shfl_xor(p1[r], 16, 32);
        }
        Frag16 bp;
        const bool lowh = (half == 0);
#pragma unroll
        for (int v = 0; v < 4; ++v) {
            float lo0 = lowh ? p0[2 * v]      : p1x[2 * v];
            float lo1 = lowh ? p0[2 * v + 1]  : p1x[2 * v + 1];
            float hi0 = lowh ? p0x[2 * v]     : p1[2 * v];
            float hi1 = lowh ? p0x[2 * v + 1] : p1[2 * v + 1];
            bp.u[v]     = pack_bf16(lo0, lo1);
            bp.u[4 + v] = pack_bf16(hi0, hi1);
        }
        // O^T += V^T . P^T   (A = V^T rows are d, K-dim = s, contiguous loads)
#pragma unroll
        for (int i = 0; i < 4; ++i) {
            v16bf av = load_a_bf16(Vn, T_SEQ, i * 16, s0, lane);
            o[i] = wmma_bf16(av, bp.v, o[i]);
        }
    }

    const float inv = 1.0f / l_run;
    // store to out-proj input layout: O[m=t*B+b][e=h*64+d], bf16
    bf16_t* orow = Obuf + ((size_t)t * BATCH + bb) * EMB + h * HD;
#pragma unroll
    for (int i = 0; i < 4; ++i) {
        uint4 w;
        w.x = pack_bf16(o[i][0] * inv, o[i][1] * inv);
        w.y = pack_bf16(o[i][2] * inv, o[i][3] * inv);
        w.z = pack_bf16(o[i][4] * inv, o[i][5] * inv);
        w.w = pack_bf16(o[i][6] * inv, o[i][7] * inv);
        *(uint4*)(orow + i * 16 + half * 8) = w;
    }
}

// ---------------------------------------------------------------------------
// Kernel 3: output projection. Y[m,f] = O[m,:] . Wout[f,:] + bout[f]
// M=4096, N=1024, K=1024; Y is d_out ([T,B,E], m = t*B+b matches flat order).
// ---------------------------------------------------------------------------
__global__ void __launch_bounds__(256)
out_proj_kernel(const bf16_t* __restrict__ O, const bf16_t* __restrict__ W,
                const float* __restrict__ bias, float* __restrict__ Y) {
    const int lane = threadIdx.x & 31;
    const int wid  = blockIdx.x * (blockDim.x >> 5) + (threadIdx.x >> 5);
    const int mtile = wid >> 4;       // 0..255
    const int ngrp  = wid & 15;       // 0..15
    const int m0 = mtile * 16;
    const int f0 = ngrp * 64;

    v8f acc[4] = {};
    for (int k0 = 0; k0 < EMB; k0 += 32) {
        v16bf a = load_a_bf16(O, EMB, m0, k0, lane);
#pragma unroll
        for (int j = 0; j < 4; ++j) {
            v16bf b = load_b_bf16(W, EMB, f0 + j * 16, k0, lane);
            acc[j] = wmma_bf16(a, b, acc[j]);
        }
    }
    const int half = lane >> 4, lr = lane & 15;
#pragma unroll
    for (int j = 0; j < 4; ++j) {
        const int f  = f0 + j * 16 + lr;
        const float bv = bias[f];
#pragma unroll
        for (int r = 0; r < 8; ++r) {
            const int m = m0 + half * 8 + r;
            Y[(size_t)m * EMB + f] = acc[j][r] + bv;
        }
    }
}

// ---------------------------------------------------------------------------
extern "C" void kernel_launch(void* const* d_in, const int* in_sizes, int n_in,
                              void* d_out, int out_size, void* d_ws, size_t ws_size,
                              hipStream_t stream) {
    const float* query = (const float*)d_in[0];          // [T,B,E]
    const float* pbias = (const float*)d_in[1];          // [B*H,T,T]
    const unsigned char* mask = (const unsigned char*)d_in[2]; // [B,T] bool
    const float* Wqkv = (const float*)d_in[3];           // [3E,E]
    const float* bqkv = (const float*)d_in[4];           // [3E]
    const float* Wout = (const float*)d_in[5];           // [E,E]
    const float* bout = (const float*)d_in[6];           // [E]
    float* Y = (float*)d_out;                            // [T,B,E]

    const size_t headElems = (size_t)NHEAD * T_SEQ * HD; // 4 Mi elems
    bf16_t* Qb    = (bf16_t*)d_ws;
    bf16_t* Kb    = Qb + headElems;                      //  +4 Mi
    bf16_t* Vt    = Kb + headElems;                      //  +4 Mi
    bf16_t* Ob    = Vt + headElems;                      //  +4 Mi  [4096,1024]
    bf16_t* Xbf   = Ob + (size_t)4096 * EMB;             //  +4 Mi  [4096,1024]
    bf16_t* Wqkvb = Xbf + (size_t)4096 * EMB;            //  +3 Mi  [3072,1024]
    bf16_t* Woutb = Wqkvb + (size_t)3072 * EMB;          //  +1 Mi  [1024,1024]

    // pre-convert activations + weights to bf16 (one pass; lives in L2)
    cvt_bf16_kernel<<<2048, 256, 0, stream>>>(query, Xbf, 4096 * EMB / 8);
    cvt_bf16_kernel<<<1536, 256, 0, stream>>>(Wqkv, Wqkvb, 3072 * EMB / 8);
    cvt_bf16_kernel<<< 512, 256, 0, stream>>>(Wout, Woutb, 1024 * EMB / 8);

    // 12288 waves = 256 Mtiles x 48 Ngroups; 8 waves/block
    qkv_proj_kernel<<<1536, 256, 0, stream>>>(Xbf, Wqkvb, bqkv, Qb, Kb, Vt);

    dim3 g2(NHEAD, T_SEQ / 16);                          // 64 x 64 single-wave WGs
    attention_kernel<<<g2, 32, 0, stream>>>(Qb, Kb, Vt, pbias, mask, Ob);

    // 4096 waves = 256 Mtiles x 16 Ngroups; 8 waves/block
    out_proj_kernel<<<512, 256, 0, stream>>>(Ob, Woutb, bout, Y);
}
